// FCCaps_40140764348532
// MI455X (gfx1250) — matrix operand order repaired and checked by reading, not compiled, not run on hardware
//
#include <hip/hip_runtime.h>

#define B_     32
#define NIN_   512
#define NOUT_  128
#define CIN_   128
#define COUT_  32
#define NITER_ 10

typedef __bf16 bf16_t;
typedef __attribute__((ext_vector_type(16))) __bf16 v16bf;
typedef __attribute__((ext_vector_type(8)))  __bf16 v8bf;
typedef __attribute__((ext_vector_type(8)))  float  v8f;

static __device__ __forceinline__ bf16_t f2bf(float f) {
    unsigned u = __builtin_bit_cast(unsigned, f);
    unsigned r = u + 0x7FFFu + ((u >> 16) & 1u);   // round-to-nearest-even
    unsigned short h = (unsigned short)(r >> 16);
    return __builtin_bit_cast(bf16_t, h);
}
static __device__ __forceinline__ float bfu2f(unsigned short h) {
    unsigned u = ((unsigned)h) << 16;
    return __builtin_bit_cast(float, u);
}

// ---------------------------------------------------------------------------
// u_hat[b][m][n][o] (bf16) = sum_c x[b][n][c] * W1[n][m][o][c]
// block = (n, m-chunk of 16); 4 waves, each owns one 16x16 tile of the 32x32 D.
// ---------------------------------------------------------------------------
__global__ __launch_bounds__(128)
void uhat_gemm(const float* __restrict__ x, const float* __restrict__ W1,
               unsigned short* __restrict__ uhat) {
    __shared__ bf16_t Xs[32 * CIN_];   // A: rows b, cols c
    __shared__ bf16_t Ws[32 * CIN_];   // B: rows o, cols c
    const int n    = blockIdx.x;
    const int mc   = blockIdx.y;
    const int tid  = threadIdx.x;
    const int lane = tid & 31;
    const int wave = tid >> 5;

    // Stage X tile fp32 -> bf16 (shared by all 16 m's of this block)
    {
        const int bb = tid >> 2, q = tid & 3;
        const float4* src = (const float4*)(x + ((size_t)bb * NIN_ + n) * CIN_ + q * 32);
        bf16_t* dst = &Xs[bb * CIN_ + q * 32];
        #pragma unroll
        for (int j = 0; j < 8; ++j) {
            float4 f = src[j];
            dst[j*4+0] = f2bf(f.x); dst[j*4+1] = f2bf(f.y);
            dst[j*4+2] = f2bf(f.z); dst[j*4+3] = f2bf(f.w);
        }
    }

    const int rt    = wave >> 1;          // D row tile (b)
    const int ct    = wave & 1;           // D col tile (o)
    const int l15   = lane & 15;
    const int hi    = lane >> 4;          // lanes 16-31
    const int arow  = rt * 16 + l15;      // A row (M)
    const int abase = hi * 16;            // A k-phase: contiguous k = 16*hi + s
    const int ocol  = ct * 16 + l15;      // B column (N)
    const int koff  = hi * 16;            // B k-phase: contiguous k = 16*hi + s

    for (int mi = 0; mi < 16; ++mi) {
        const int m = mc * 16 + mi;
        __syncthreads();                  // Ws reuse fence
        {   // Stage W tile fp32 -> bf16
            const int oo = tid >> 2, q = tid & 3;
            const float4* src =
                (const float4*)(W1 + (((size_t)n * NOUT_ + m) * COUT_ + oo) * CIN_ + q * 32);
            bf16_t* dst = &Ws[oo * CIN_ + q * 32];
            #pragma unroll
            for (int j = 0; j < 8; ++j) {
                float4 f = src[j];
                dst[j*4+0] = f2bf(f.x); dst[j*4+1] = f2bf(f.y);
                dst[j*4+2] = f2bf(f.z); dst[j*4+3] = f2bf(f.w);
            }
        }
        __syncthreads();

        v8f acc = {};
        #pragma unroll
        for (int kk = 0; kk < 4; ++kk) {
            const int k0 = kk * 32;
            v8bf alo = *(const v8bf*)&Xs[arow * CIN_ + k0 + abase];
            v8bf ahi = *(const v8bf*)&Xs[arow * CIN_ + k0 + abase + 8];
            v16bf afrag;
            #pragma unroll
            for (int i = 0; i < 8; ++i) { afrag[i] = alo[i]; afrag[i + 8] = ahi[i]; }
            v16bf bfrag = *(const v16bf*)&Ws[ocol * CIN_ + k0 + koff];
            acc = __builtin_amdgcn_wmma_f32_16x16x32_bf16(
                false, afrag, false, bfrag, (short)0, acc, false, false);
        }

        // D layout: VGPR r -> M=r (lanes 0-15) / M=r+8 (lanes 16-31), N=l15
        #pragma unroll
        for (int r = 0; r < 8; ++r) {
            const int bb = rt * 16 + hi * 8 + r;
            const size_t idx = (((size_t)bb * NOUT_ + m) * NIN_ + n) * COUT_ + ocol;
            uhat[idx] = __builtin_bit_cast(unsigned short, f2bf(acc[r]));
        }
    }
}

// ---------------------------------------------------------------------------
// c[b][n][m] = softmax_m(blog[b][n][m]); one 128-thread block per (b,n) row
// ---------------------------------------------------------------------------
__global__ __launch_bounds__(128)
void route_softmax(const float* __restrict__ blog, float* __restrict__ c) {
    const int row = blockIdx.x;           // b*NIN + n
    const int m   = threadIdx.x;
    __shared__ float red[128];
    const float v = blog[(size_t)row * NOUT_ + m];
    red[m] = v; __syncthreads();
    for (int s = 64; s > 0; s >>= 1) {
        if (m < s) red[m] = fmaxf(red[m], red[m + s]);
        __syncthreads();
    }
    const float mx = red[0]; __syncthreads();
    const float e = __expf(v - mx);
    red[m] = e; __syncthreads();
    for (int s = 64; s > 0; s >>= 1) {
        if (m < s) red[m] = red[m] + red[m + s];
        __syncthreads();
    }
    c[(size_t)row * NOUT_ + m] = e / red[0];
}

// ---------------------------------------------------------------------------
// s[b][m][o] = sum_n c[b][n][m] * uhat[b][m][n][o];  v = (|s|/(0.5+s^2))*s
// c==null -> uniform 1/NOUT (iteration 0). poses/acts non-null on final iter.
// block per (b,m): 8 n-groups x 32 o lanes, fixed-order reduction.
// ---------------------------------------------------------------------------
__global__ __launch_bounds__(256)
void route_sv(const unsigned short* __restrict__ uhat, const float* __restrict__ c,
              float* __restrict__ v, float* __restrict__ poses, float* __restrict__ acts) {
    const int bm = blockIdx.x;            // b*NOUT + m
    const int b  = bm >> 7;
    const int m  = bm & 127;
    const int o  = threadIdx.x & 31;
    const int g  = threadIdx.x >> 5;
    __shared__ float red[8][32];
    float acc = 0.f;
    const size_t base = (size_t)bm * NIN_ * COUT_;
    for (int n = g; n < NIN_; n += 8) {
        const float cw = c ? c[((size_t)b * NIN_ + n) * NOUT_ + m] : (1.0f / NOUT_);
        acc += cw * bfu2f(uhat[base + (size_t)n * COUT_ + o]);
    }
    red[g][o] = acc;
    __syncthreads();
    if (g == 0) {
        float s = red[0][o];
        #pragma unroll
        for (int i = 1; i < 8; ++i) s += red[i][o];
        const float s2 = s * s;
        const float vv = (sqrtf(s2) / (0.5f + s2)) * s;
        v[(size_t)bm * COUT_ + o] = vv;
        if (poses) poses[(size_t)bm * COUT_ + o] = vv;
        red[0][o] = vv * vv;
    }
    if (acts) {
        __syncthreads();
        if (threadIdx.x == 0) {
            float t = 0.f;
            #pragma unroll
            for (int i = 0; i < 32; ++i) t += red[0][i];
            acts[bm] = sqrtf(t);
        }
    }
}

// ---------------------------------------------------------------------------
// blog[b][n][m] = (addOld? old:0) + sum_o v[b][m][o]*uhat[b][m][n][o]
// exactly B*NIN*NOUT threads; 64B vectorized uhat reads (L2-resident)
// ---------------------------------------------------------------------------
__global__ __launch_bounds__(256)
void route_agree(const unsigned short* __restrict__ uhat, const float* __restrict__ v,
                 float* __restrict__ blog, int addOld) {
    const int gid = blockIdx.x * 256 + threadIdx.x;   // (b*NIN+n)*NOUT + m
    const int m  = gid & 127;
    const int bn = gid >> 7;
    const int n  = bn & 511;
    const int b  = bn >> 9;
    const size_t ub = (((size_t)b * NOUT_ + m) * NIN_ + n) * COUT_;
    const float* vp = v + ((size_t)b * NOUT_ + m) * COUT_;
    const uint4* up = (const uint4*)(uhat + ub);
    float dot = 0.f;
    #pragma unroll
    for (int j = 0; j < 4; ++j) {
        const uint4 u = up[j];
        const unsigned arr[4] = {u.x, u.y, u.z, u.w};
        #pragma unroll
        for (int k = 0; k < 4; ++k) {
            const float f0 = __builtin_bit_cast(float, (arr[k] & 0xFFFFu) << 16);
            const float f1 = __builtin_bit_cast(float, arr[k] & 0xFFFF0000u);
            dot += vp[j * 8 + k * 2 + 0] * f0;
            dot += vp[j * 8 + k * 2 + 1] * f1;
        }
    }
    blog[gid] = (addOld ? blog[gid] : 0.f) + dot;
}

// ---------------------------------------------------------------------------
extern "C" void kernel_launch(void* const* d_in, const int* in_sizes, int n_in,
                              void* d_out, int out_size, void* d_ws, size_t ws_size,
                              hipStream_t stream) {
    (void)in_sizes; (void)n_in; (void)out_size; (void)ws_size;
    const float* x  = (const float*)d_in[0];
    const float* W1 = (const float*)d_in[1];
    // d_in[2] = y, unused by forward

    char* ws = (char*)d_ws;
    unsigned short* uhat = (unsigned short*)ws;                                 // 128 MiB
    size_t off = (size_t)B_ * NOUT_ * NIN_ * COUT_ * sizeof(unsigned short);
    float* blog = (float*)(ws + off); off += (size_t)B_ * NIN_ * NOUT_ * sizeof(float);
    float* cbuf = (float*)(ws + off); off += (size_t)B_ * NIN_ * NOUT_ * sizeof(float);
    float* vbuf = (float*)(ws + off);

    float* poses = (float*)d_out;
    float* acts  = poses + (size_t)B_ * NOUT_ * COUT_;

    // 1) u_hat via bf16 WMMA (HBM-bound on W1; compute far under the memory floor)
    dim3 gg(NIN_, NOUT_ / 16);
    uhat_gemm<<<gg, 128, 0, stream>>>(x, W1, uhat);

    // 2) routing step 1: b==0 -> softmax is uniform 1/NOUT; blog written fresh
    route_sv<<<B_ * NOUT_, 256, 0, stream>>>(uhat, nullptr, vbuf, nullptr, nullptr);
    route_agree<<<(B_ * NIN_ * NOUT_) / 256, 256, 0, stream>>>(uhat, vbuf, blog, 0);

    // 3) routing steps 2..9
    for (int it = 1; it < NITER_ - 1; ++it) {
        route_softmax<<<B_ * NIN_, 128, 0, stream>>>(blog, cbuf);
        route_sv<<<B_ * NOUT_, 256, 0, stream>>>(uhat, cbuf, vbuf, nullptr, nullptr);
        route_agree<<<(B_ * NIN_ * NOUT_) / 256, 256, 0, stream>>>(uhat, vbuf, blog, 1);
    }

    // 4) final step 10: v -> poses, activations; final b-update is discarded
    route_softmax<<<B_ * NIN_, 128, 0, stream>>>(blog, cbuf);
    route_sv<<<B_ * NOUT_, 256, 0, stream>>>(uhat, cbuf, vbuf, poses, acts);
}
